// GATGNN_36240934043736
// MI455X (gfx1250) — compile-verified, matrix operand order checked by reading, not run on hardware
//
#include <hip/hip_runtime.h>
#include <math.h>

// ---------------------------------------------------------------------------
// Types for CDNA5 WMMA (gfx1250, wave32)
// ---------------------------------------------------------------------------
typedef __bf16 bf16;
typedef __attribute__((ext_vector_type(16))) __bf16 v16bf;
typedef __attribute__((ext_vector_type(8)))  __bf16 v8bf;
typedef __attribute__((ext_vector_type(8)))  float  v8f;

static const int N1_ = 200000, N2_ = 100000, N3_ = 50000;
static const int E1_ = 1200000, E2_ = 600000, E3_ = 300000;
static const int NH = 2; // heads

#define GRID1(n) ((unsigned)(((n) + 255) / 256))

// ---------------------------------------------------------------------------
// Small helpers
// ---------------------------------------------------------------------------
__device__ inline void atomicMaxF(float* addr, float v) {
  // classic sign-split trick; correct for mixed signs, init to -inf
  if (v >= 0.0f) atomicMax((int*)addr, __float_as_int(v));
  else           atomicMin((unsigned int*)addr, __float_as_uint(v));
}

__global__ void k_fill(float* __restrict__ p, float v, size_t n) {
  size_t i = (size_t)blockIdx.x * blockDim.x + threadIdx.x;
  if (i < n) p[i] = v;
}

__global__ void k_cast_bf16(const float* __restrict__ s, bf16* __restrict__ d, size_t n) {
  size_t i = (size_t)blockIdx.x * blockDim.x + threadIdx.x;
  if (i < n) d[i] = (bf16)s[i];
}

// W [K,N] f32 row-major  ->  Wt [N,K] bf16 row-major (transposed for B-fragment loads)
__global__ void k_wcast_t(const float* __restrict__ W, bf16* __restrict__ Wt, int K, int N) {
  int i = blockIdx.x * blockDim.x + threadIdx.x;
  if (i >= K * N) return;
  int k = i / N, n = i % N;
  Wt[(size_t)n * K + k] = (bf16)W[i];
}

// ---------------------------------------------------------------------------
// WMMA bf16 GEMM:  C[M,N] = A[M,K] (bf16) * Bt[N,K]^T (bf16)  (+bias, +relu)
// one wave per 16x16 output tile; M,N multiples of 16; K multiple of 32.
// A-fragment layout (16-bit, 16x32): lane row = lane&15,
//   element e -> K = (e>>3)*16 + (lane>>4)*8 + (e&7).  B symmetric (col = lane&15).
// C/D: col = lane&15, row = v + 8*(lane>>4).
// ---------------------------------------------------------------------------
__global__ void __launch_bounds__(256) k_gemm_bf16(
    const bf16* __restrict__ A, const bf16* __restrict__ Bt,
    const float* __restrict__ bias, float* __restrict__ C,
    int M, int N, int K, int ldc, int act)
{
  int wave = threadIdx.x >> 5;
  int lane = threadIdx.x & 31;
  int mt = M >> 4, nt = N >> 4;
  int tile = blockIdx.x * 8 + wave;
  if (tile >= mt * nt) return;            // wave-uniform exit: EXEC stays all-1s
  int m0 = (tile % mt) << 4;
  int n0 = (tile / mt) << 4;
  int half = lane >> 4;
  int l15  = lane & 15;

  const bf16* arow = A  + (size_t)(m0 + l15) * K + half * 8;
  const bf16* brow = Bt + (size_t)(n0 + l15) * K + half * 8;

  v8f acc = {};
  for (int k0 = 0; k0 < K; k0 += 32) {
    v8bf alo = *(const v8bf*)(arow + k0);
    v8bf ahi = *(const v8bf*)(arow + k0 + 16);
    v8bf blo = *(const v8bf*)(brow + k0);
    v8bf bhi = *(const v8bf*)(brow + k0 + 16);
    v16bf a, b;
#pragma unroll
    for (int i = 0; i < 8; ++i) {
      a[i] = alo[i]; a[8 + i] = ahi[i];
      b[i] = blo[i]; b[8 + i] = bhi[i];
    }
    acc = __builtin_amdgcn_wmma_f32_16x16x32_bf16(
        false, a, false, b, (short)0, acc, false, false);
  }

  int col = n0 + l15;
  float bv = bias ? bias[col] : 0.0f;
#pragma unroll
  for (int v = 0; v < 8; ++v) {
    float r = acc[v] + bv;
    if (act == 1) r = r > 0.0f ? r : 0.0f;
    C[(size_t)(m0 + v + 8 * half) * ldc + col] = r;
  }
}

// conv1 (K=6): plain f32 GEMM, one thread per output element
__global__ void k_gemm_small(const float* __restrict__ x, const float* __restrict__ W,
                             float* __restrict__ h, int N, int K, int Nn) {
  size_t i = (size_t)blockIdx.x * blockDim.x + threadIdx.x;
  if (i >= (size_t)N * Nn) return;
  int n = (int)(i / Nn), j = (int)(i % Nn);
  float s = 0.0f;
  for (int k = 0; k < K; ++k) s += x[(size_t)n * K + k] * W[k * Nn + j];
  h[i] = s;
}

// ---------------------------------------------------------------------------
// GAT pieces
// ---------------------------------------------------------------------------
// a_src[n,h] = sum_c h[n,h,c]*att_src[h,c]; same for a_dst
__global__ void k_node_att(const float* __restrict__ h, const float* __restrict__ attS,
                           const float* __restrict__ attD, float* __restrict__ asrc,
                           float* __restrict__ adst, int N, int Cc) {
  size_t i = (size_t)blockIdx.x * blockDim.x + threadIdx.x;
  if (i >= (size_t)N * NH) return;
  int n = (int)(i / NH), hh = (int)(i % NH);
  const float* hp = h + (size_t)n * NH * Cc + hh * Cc;
  float s1 = 0.0f, s2 = 0.0f;
  for (int c = 0; c < Cc; ++c) {
    float v = hp[c];
    s1 += v * attS[hh * Cc + c];
    s2 += v * attD[hh * Cc + c];
  }
  asrc[i] = s1; adst[i] = s2;
}

// pass 1: e = leaky_relu(a_src[s]+a_dst[d]); segment atomic max into emax[d]
__global__ void k_edge_emax(const int* __restrict__ src, const int* __restrict__ dst,
                            int E, int N, const float* __restrict__ asrc,
                            const float* __restrict__ adst, float* __restrict__ ebuf,
                            float* __restrict__ emax) {
  size_t i = (size_t)blockIdx.x * blockDim.x + threadIdx.x;
  size_t total = (size_t)(E + N) * NH;
  if (i >= total) return;
  int e = (int)(i / NH), h = (int)(i % NH);
  int s, d;
  if (e < E) { s = src[e]; d = dst[e]; } else { s = d = e - E; }
  float v = asrc[s * NH + h] + adst[d * NH + h];
  v = v > 0.0f ? v : 0.2f * v;
  ebuf[i] = v;
  atomicMaxF(&emax[d * NH + h], v);
}

__global__ void k_emax_fix(float* __restrict__ emax, size_t n) {
  size_t i = (size_t)blockIdx.x * blockDim.x + threadIdx.x;
  if (i >= n) return;
  float v = emax[i];
  if (!(v > -3.0e38f && v < 3.0e38f)) emax[i] = 0.0f;  // non-finite -> 0
}

// pass 2: ex = exp(e - emax[d]); segment atomic sum into denom[d]
__global__ void k_edge_expsum(const int* __restrict__ dst, int E, int N,
                              const float* __restrict__ emax, float* __restrict__ ebuf,
                              float* __restrict__ denom) {
  size_t i = (size_t)blockIdx.x * blockDim.x + threadIdx.x;
  size_t total = (size_t)(E + N) * NH;
  if (i >= total) return;
  int e = (int)(i / NH), h = (int)(i % NH);
  int d = (e < E) ? dst[e] : e - E;
  float ex = __expf(ebuf[i] - emax[d * NH + h]);
  ebuf[i] = ex;
  atomicAdd(&denom[d * NH + h], ex);
}

// pass 3: out[d, f] += (ex/denom[d,h]) * hfeat[s, f]
__global__ void k_edge_agg(const int* __restrict__ src, const int* __restrict__ dst,
                           int E, int N, const float* __restrict__ ebuf,
                           const float* __restrict__ denom, const float* __restrict__ hfeat,
                           float* __restrict__ out, int Cc, int HC) {
  size_t i = (size_t)blockIdx.x * blockDim.x + threadIdx.x;
  size_t total = (size_t)(E + N) * HC;
  if (i >= total) return;
  int e = (int)(i / HC), f = (int)(i % HC), h = f / Cc;
  int s, d;
  if (e < E) { s = src[e]; d = dst[e]; } else { s = d = e - E; }
  float w = ebuf[(size_t)e * NH + h] / fmaxf(denom[d * NH + h], 1e-16f);
  atomicAdd(&out[(size_t)d * HC + f], w * hfeat[(size_t)s * HC + f]);
}

__global__ void k_bias(float* __restrict__ out, const float* __restrict__ bias,
                       size_t n, int F) {
  size_t i = (size_t)blockIdx.x * blockDim.x + threadIdx.x;
  if (i < n) out[i] += bias[i % F];
}

// ---------------------------------------------------------------------------
// Pool / unpool / concat
// ---------------------------------------------------------------------------
__global__ void k_pool_scatter(const float* __restrict__ x, const int* __restrict__ cl,
                               float* __restrict__ sum, float* __restrict__ cnt,
                               int N, int C) {
  size_t i = (size_t)blockIdx.x * blockDim.x + threadIdx.x;
  if (i >= (size_t)N * C) return;
  int n = (int)(i / C), c = (int)(i % C);
  int g = cl[n];
  atomicAdd(&sum[(size_t)g * C + c], x[i]);
  if (c == 0) atomicAdd(&cnt[g], 1.0f);
}

__global__ void k_pool_div(float* __restrict__ p, const float* __restrict__ cnt,
                           size_t n, int C) {
  size_t i = (size_t)blockIdx.x * blockDim.x + threadIdx.x;
  if (i < n) p[i] /= fmaxf(cnt[i / C], 1.0f);
}

__global__ void k_gather(const float* __restrict__ src, const int* __restrict__ idx,
                         float* __restrict__ dst, int N, int C) {
  size_t i = (size_t)blockIdx.x * blockDim.x + threadIdx.x;
  if (i >= (size_t)N * C) return;
  int n = (int)(i / C), c = (int)(i % C);
  dst[i] = src[(size_t)idx[n] * C + c];
}

__global__ void k_copycols(const float* __restrict__ src, float* __restrict__ dst,
                           int N, int C, int ld, int off) {
  size_t i = (size_t)blockIdx.x * blockDim.x + threadIdx.x;
  if (i >= (size_t)N * C) return;
  int n = (int)(i / C), c = (int)(i % C);
  dst[(size_t)n * ld + off + c] = src[i];
}

// fc3 (128->3) + tanh + L2 normalize
__global__ void k_fc3(const float* __restrict__ x, const float* __restrict__ W,
                      const float* __restrict__ b, float* __restrict__ out, int N) {
  int n = blockIdx.x * blockDim.x + threadIdx.x;
  if (n >= N) return;
  const float* xp = x + (size_t)n * 128;
  float o[3];
  for (int j = 0; j < 3; ++j) {
    float s = b[j];
    for (int k = 0; k < 128; ++k) s += xp[k] * W[k * 3 + j];
    o[j] = tanhf(s);
  }
  float nrm = fmaxf(sqrtf(o[0]*o[0] + o[1]*o[1] + o[2]*o[2]), 1e-12f);
  out[(size_t)n * 3 + 0] = o[0] / nrm;
  out[(size_t)n * 3 + 1] = o[1] / nrm;
  out[(size_t)n * 3 + 2] = o[2] / nrm;
}

// ---------------------------------------------------------------------------
// Driver
// ---------------------------------------------------------------------------
extern "C" void kernel_launch(void* const* d_in, const int* in_sizes, int n_in,
                              void* d_out, int out_size, void* d_ws, size_t ws_size,
                              hipStream_t stream) {
  (void)in_sizes; (void)n_in; (void)out_size; (void)ws_size;

  const float* x   = (const float*)d_in[0];
  const int*   ei1 = (const int*)d_in[1];
  const int*   ei2 = (const int*)d_in[2];
  const int*   ei3 = (const int*)d_in[3];
  const int*   c2  = (const int*)d_in[4];
  const int*   c3  = (const int*)d_in[5];
  int pi = 6;
  auto P = [&]() { return (const float*)d_in[pi++]; };
  // params in dict order: conv1..conv4, lin5, conv5, lin6, conv6, fc1, fc2, fc3
  const float *W1 = P(), *aS1 = P(), *aD1 = P(), *b1 = P();
  const float *W2 = P(), *aS2 = P(), *aD2 = P(), *b2 = P();
  const float *W3 = P(), *aS3 = P(), *aD3 = P(), *b3 = P();
  const float *W4 = P(), *aS4 = P(), *aD4 = P(), *b4 = P();
  const float *W5l = P(), *aS5l = P(), *aD5l = P(), *b5l = P();
  const float *W5 = P(), *aS5 = P(), *aD5 = P(), *b5 = P();
  const float *W6l = P(), *aS6l = P(), *aD6l = P(), *b6l = P();
  const float *W6 = P(), *aS6 = P(), *aD6 = P(), *b6 = P();
  const float *Wf1 = P(), *bf1 = P();
  const float *Wf2 = P(), *bf2 = P();
  const float *Wf3 = P(), *bf3 = P();

  // ---- workspace bump allocator ----
  char* wsp = (char*)d_ws;
  size_t off = 0;
  auto alloc = [&](size_t bytes) -> void* {
    void* p = wsp + off;
    off += (bytes + 255) & ~(size_t)255;
    return p;
  };
  bf16* wt2  = (bf16*)alloc((size_t)64 * 128 * 2);
  bf16* wt3  = (bf16*)alloc((size_t)128 * 256 * 2);
  bf16* wt4  = (bf16*)alloc((size_t)256 * 256 * 2);
  bf16* wt5l = (bf16*)alloc((size_t)256 * 128 * 2);
  bf16* wt5  = (bf16*)alloc((size_t)256 * 128 * 2);
  bf16* wt6l = (bf16*)alloc((size_t)128 * 64 * 2);
  bf16* wt6  = (bf16*)alloc((size_t)128 * 64 * 2);
  bf16* wtf1 = (bf16*)alloc((size_t)64 * 512 * 2);
  bf16* wtf2 = (bf16*)alloc((size_t)512 * 128 * 2);

  float* hbuf  = (float*)alloc((size_t)12800000 * 4);          // max N*HC (51.2 MB)
  bf16*  stage = (bf16*)alloc((size_t)N1_ * 512 * 2);          // bf16 staging (204.8 MB)
  float* ebuf  = (float*)alloc((size_t)(E1_ + N1_) * NH * 4);  // edge scratch
  float* asrc  = (float*)alloc((size_t)N1_ * NH * 4);
  float* adst  = (float*)alloc((size_t)N1_ * NH * 4);
  float* emaxb = (float*)alloc((size_t)N1_ * NH * 4);
  float* denom = (float*)alloc((size_t)N1_ * NH * 4);
  float* cnt   = (float*)alloc((size_t)N2_ * 4);
  float* pool  = (float*)alloc((size_t)N2_ * 64 * 4);          // == N3*128
  float* x1    = (float*)alloc((size_t)N1_ * 64 * 4);
  float* x2    = (float*)alloc((size_t)N2_ * 128 * 4);
  float* x3    = (float*)alloc((size_t)N3_ * 256 * 4);
  float* mid5  = (float*)alloc((size_t)N2_ * 128 * 4);
  float* mid5b = (float*)alloc((size_t)N2_ * 128 * 4);
  float* mid6  = (float*)alloc((size_t)N1_ * 64 * 4);
  float* fbuf  = (float*)alloc((size_t)N1_ * 64 * 4);
  float* tmpA  = (float*)alloc((size_t)N1_ * 512 * 4);         // 409.6 MB multi-use
  float* f2din = tmpA;                    // [N2,256]
  float* cat2  = tmpA + (size_t)51200000; // [N2,256] (upper region)
  float* f1din = tmpA;                    // [N1,128]
  float* cat1  = tmpA + (size_t)51200000; // [N1,128]

  auto gemm = [&](const bf16* A, const bf16* Bt, const float* bias, float* C,
                  int M, int Nn, int K, int ldc, int act) {
    int tiles = (M >> 4) * (Nn >> 4);
    k_gemm_bf16<<<(unsigned)((tiles + 7) / 8), 256, 0, stream>>>(A, Bt, bias, C,
                                                                 M, Nn, K, ldc, act);
  };

  // Generic GAT layer (PyG-style, self-loops, softmax per dst, concat heads)
  auto run_gat = [&](const float* xin, int N, int Cin,
                     const int* esrc, const int* edst, int E,
                     const bf16* Wt, const float* Wf32,
                     const float* attS, const float* attD, const float* bias,
                     int Cc, float* out) {
    int HC = NH * Cc;
    size_t nHC = (size_t)N * HC;
    if (Wt) {
      size_t nin = (size_t)N * Cin;
      k_cast_bf16<<<GRID1(nin), 256, 0, stream>>>(xin, stage, nin);
      gemm(stage, Wt, nullptr, hbuf, N, HC, Cin, HC, 0);
    } else {
      k_gemm_small<<<GRID1(nHC), 256, 0, stream>>>(xin, Wf32, hbuf, N, Cin, HC);
    }
    size_t nNH = (size_t)N * NH;
    k_node_att<<<GRID1(nNH), 256, 0, stream>>>(hbuf, attS, attD, asrc, adst, N, Cc);
    k_fill<<<GRID1(nNH), 256, 0, stream>>>(emaxb, -__builtin_huge_valf(), nNH);
    k_fill<<<GRID1(nNH), 256, 0, stream>>>(denom, 0.0f, nNH);
    k_fill<<<GRID1(nHC), 256, 0, stream>>>(out, 0.0f, nHC);
    size_t eT = (size_t)(E + N) * NH;
    k_edge_emax<<<GRID1(eT), 256, 0, stream>>>(esrc, edst, E, N, asrc, adst, ebuf, emaxb);
    k_emax_fix<<<GRID1(nNH), 256, 0, stream>>>(emaxb, nNH);
    k_edge_expsum<<<GRID1(eT), 256, 0, stream>>>(edst, E, N, emaxb, ebuf, denom);
    size_t aT = (size_t)(E + N) * HC;
    k_edge_agg<<<GRID1(aT), 256, 0, stream>>>(esrc, edst, E, N, ebuf, denom, hbuf,
                                              out, Cc, HC);
    k_bias<<<GRID1(nHC), 256, 0, stream>>>(out, bias, nHC, HC);
  };

  // ---- weight prep (cast + transpose to bf16) ----
  k_wcast_t<<<GRID1(64 * 128), 256, 0, stream>>>(W2, wt2, 64, 128);
  k_wcast_t<<<GRID1(128 * 256), 256, 0, stream>>>(W3, wt3, 128, 256);
  k_wcast_t<<<GRID1(256 * 256), 256, 0, stream>>>(W4, wt4, 256, 256);
  k_wcast_t<<<GRID1(256 * 128), 256, 0, stream>>>(W5l, wt5l, 256, 128);
  k_wcast_t<<<GRID1(256 * 128), 256, 0, stream>>>(W5, wt5, 256, 128);
  k_wcast_t<<<GRID1(128 * 64), 256, 0, stream>>>(W6l, wt6l, 128, 64);
  k_wcast_t<<<GRID1(128 * 64), 256, 0, stream>>>(W6, wt6, 128, 64);
  k_wcast_t<<<GRID1(64 * 512), 256, 0, stream>>>(Wf1, wtf1, 64, 512);
  k_wcast_t<<<GRID1(512 * 128), 256, 0, stream>>>(Wf2, wtf2, 512, 128);

  // ---- pipeline ----
  // conv1 (K=6, scalar GEMM path)
  run_gat(x, N1_, 6, ei1, ei1 + E1_, E1_, nullptr, W1, aS1, aD1, b1, 32, x1);

  // pool r1->r2
  k_fill<<<GRID1((size_t)N2_ * 64), 256, 0, stream>>>(pool, 0.0f, (size_t)N2_ * 64);
  k_fill<<<GRID1((size_t)N2_), 256, 0, stream>>>(cnt, 0.0f, (size_t)N2_);
  k_pool_scatter<<<GRID1((size_t)N1_ * 64), 256, 0, stream>>>(x1, c2, pool, cnt, N1_, 64);
  k_pool_div<<<GRID1((size_t)N2_ * 64), 256, 0, stream>>>(pool, cnt, (size_t)N2_ * 64, 64);

  run_gat(pool, N2_, 64, ei2, ei2 + E2_, E2_, wt2, nullptr, aS2, aD2, b2, 64, x2);

  // pool r2->r3
  k_fill<<<GRID1((size_t)N3_ * 128), 256, 0, stream>>>(pool, 0.0f, (size_t)N3_ * 128);
  k_fill<<<GRID1((size_t)N3_), 256, 0, stream>>>(cnt, 0.0f, (size_t)N3_);
  k_pool_scatter<<<GRID1((size_t)N2_ * 128), 256, 0, stream>>>(x2, c3, pool, cnt, N2_, 128);
  k_pool_div<<<GRID1((size_t)N3_ * 128), 256, 0, stream>>>(pool, cnt, (size_t)N3_ * 128, 128);

  run_gat(pool, N3_, 128, ei3, ei3 + E3_, E3_, wt3, nullptr, aS3, aD3, b3, 128, x3);
  run_gat(x3, N3_, 256, ei3, ei3 + E3_, E3_, wt4, nullptr, aS4, aD4, b4, 128, x3); // in-place ok (input staged first)

  // unpool r3->r2, lin5, concat, conv5
  k_gather<<<GRID1((size_t)N2_ * 256), 256, 0, stream>>>(x3, c3, f2din, N2_, 256);
  run_gat(f2din, N2_, 256, ei2, ei2 + E2_, E2_, wt5l, nullptr, aS5l, aD5l, b5l, 64, mid5);
  k_copycols<<<GRID1((size_t)N2_ * 128), 256, 0, stream>>>(x2, cat2, N2_, 128, 256, 0);
  k_copycols<<<GRID1((size_t)N2_ * 128), 256, 0, stream>>>(mid5, cat2, N2_, 128, 256, 128);
  run_gat(cat2, N2_, 256, ei2, ei2 + E2_, E2_, wt5, nullptr, aS5, aD5, b5, 64, mid5b);

  // unpool r2->r1, lin6, concat, conv6
  k_gather<<<GRID1((size_t)N1_ * 128), 256, 0, stream>>>(mid5b, c2, f1din, N1_, 128);
  run_gat(f1din, N1_, 128, ei1, ei1 + E1_, E1_, wt6l, nullptr, aS6l, aD6l, b6l, 32, mid6);
  k_copycols<<<GRID1((size_t)N1_ * 64), 256, 0, stream>>>(x1, cat1, N1_, 64, 128, 0);
  k_copycols<<<GRID1((size_t)N1_ * 64), 256, 0, stream>>>(mid6, cat1, N1_, 64, 128, 64);
  run_gat(cat1, N1_, 128, ei1, ei1 + E1_, E1_, wt6, nullptr, aS6, aD6, b6, 32, fbuf);

  // MLP head: fc1 relu -> fc2 relu -> fc3 tanh + normalize
  k_cast_bf16<<<GRID1((size_t)N1_ * 64), 256, 0, stream>>>(fbuf, stage, (size_t)N1_ * 64);
  gemm(stage, wtf1, bf1, tmpA, N1_, 512, 64, 512, 1);
  k_cast_bf16<<<GRID1((size_t)N1_ * 512), 256, 0, stream>>>(tmpA, stage, (size_t)N1_ * 512);
  gemm(stage, wtf2, bf2, tmpA, N1_, 128, 512, 128, 1);
  k_fc3<<<GRID1((size_t)N1_), 256, 0, stream>>>(tmpA, Wf3, bf3, (float*)d_out, N1_);
}
